// DepGraph_35888746726166
// MI455X (gfx1250) — matrix operand order repaired and checked by reading, not compiled, not run on hardware
//
#include <hip/hip_runtime.h>
#include <hip/hip_bf16.h>
#include <math.h>

#define N_PTS 4096
#define DIM   256

typedef _Float16 h8   __attribute__((ext_vector_type(8)));
typedef _Float16 v16h __attribute__((ext_vector_type(16)));
typedef float    v8f  __attribute__((ext_vector_type(8)));

union H16 { h8 half8[2]; v16h h16; };

// A-fragment for v_wmma_f32_16x16x32_f16 (row-major [row][K] f16 storage):
// lane holds row (lane&15); elements 0..7 = K kbase..kbase+7, 8..15 = kbase+16..kbase+23.
__device__ __forceinline__ v16h load_frag_a(const _Float16* p) {
  H16 t;
  t.half8[0] = *reinterpret_cast<const h8*>(p);
  t.half8[1] = *reinterpret_cast<const h8*>(p + 16);
  return t.h16;
}
// B-fragment (B = uR^T conceptually, loaded from row-major [col][K]):
// lane holds col (lane&15); elements 0..7 = K kbase..kbase+7, 8..15 = kbase+16..kbase+23.
__device__ __forceinline__ v16h load_frag_b(const _Float16* p) {
  H16 t;
  t.half8[0] = *reinterpret_cast<const h8*>(p);
  t.half8[1] = *reinterpret_cast<const h8*>(p + 16);
  return t.h16;
}

__device__ __forceinline__ v8f wmma16(v16h a, v16h b, v8f c) {
  return __builtin_amdgcn_wmma_f32_16x16x32_f16(false, a, false, b, (short)0, c,
                                                false, false);
}

// logitexp + relaxed-bernoulli epilogue (matches reference math).
__device__ __forceinline__ float epi(float dot, float na, float nb, float msc, float uin) {
  const float LOG2C = 0.69314718056f;
  float d   = fmaxf(na + nb - 2.0f * dot, 0.0f);
  float lp  = msc * d;                       // -0.5*d/scale  (<= 0)
  float pos = fmaxf(lp, -LOG2C);
  float neg = fminf(lp, -LOG2C);
  float neg_val = neg - log1pf(-__expf(neg));
  float pos_val = -__logf(fmaxf(expm1f(-pos), 1e-20f));
  float logit   = pos_val + neg_val;
  float u = fminf(fmaxf(uin, 1e-6f), 1.0f - 1e-6f);
  float logistic = __logf(u) - log1pf(-u);
  float z = (logit + logistic) * (1.0f / 0.3f);   // /TEMPERATURE
  return 1.0f / (1.0f + __expf(-z));
}

// ---- per-row stats: f32->f16 convert, ||x||^2, optional sum log Phi(x) ----
__global__ void rowstats_kernel(const float* __restrict__ src,
                                _Float16* __restrict__ dst,
                                float* __restrict__ norms,
                                float* __restrict__ logcdf) {
  int row = blockIdx.x;
  int t   = threadIdx.x;           // blockDim.x == DIM == 256
  float x = src[(size_t)row * DIM + t];
  dst[(size_t)row * DIM + t] = (_Float16)x;

  __shared__ float red[DIM];
  red[t] = x * x;
  __syncthreads();
  for (int s = DIM / 2; s > 0; s >>= 1) {
    if (t < s) red[t] += red[t + s];
    __syncthreads();
  }
  if (t == 0) norms[row] = red[0];

  if (logcdf != nullptr) {         // uniform branch
    __syncthreads();
    red[t] = __logf(0.5f + 0.5f * erff(x * 0.70710678118654752f));
    __syncthreads();
    for (int s = DIM / 2; s > 0; s >>= 1) {
      if (t < s) red[t] += red[t + s];
      __syncthreads();
    }
    if (t == 0) logcdf[row] = red[0];
  }
}

// ---- rank[i] = position of i in stable ascending argsort of lc ----
__global__ void rank_kernel(const float* __restrict__ lc, int* __restrict__ rank) {
  int i   = blockIdx.x * blockDim.x + threadIdx.x;
  float v = lc[i];
  __shared__ float s[256];
  int r = 0;
  for (int base = 0; base < N_PTS; base += 256) {
    s[threadIdx.x] = lc[base + threadIdx.x];
    __syncthreads();
#pragma unroll 8
    for (int t = 0; t < 256; ++t) {
      float w = s[t];
      int   j = base + t;
      r += (w < v || (w == v && j < i)) ? 1 : 0;
    }
    __syncthreads();
  }
  rank[i] = r;
}

// ---- fused: WMMA pairwise-dist tile + logitexp + relaxed bernoulli ----
// Block = 256 threads (8 waves), block tile 64 rows x 128 cols, wave tile 32x32.
template <bool IS_G>
__global__ __launch_bounds__(256)
void pair_kernel(const _Float16* __restrict__ Af, const _Float16* __restrict__ Bf,
                 const float* __restrict__ nA, const float* __restrict__ nB,
                 const int* __restrict__ rnk, const float* __restrict__ U,
                 const float* __restrict__ g_logscale, float* __restrict__ out) {
  int lane   = threadIdx.x & 31;
  int wid    = threadIdx.x >> 5;
  int lm     = lane & 15;
  int laneHi = lane >> 4;                         // 0 or 1
  int rowBase = blockIdx.y * 64  + (wid >> 2) * 32;
  int colBase = blockIdx.x * 128 + (wid & 3) * 32;

  float msc = -0.5f * __expf(-g_logscale[0]);     // -0.5 / scale

  const _Float16* pa0 = Af + (size_t)(rowBase + lm) * DIM + laneHi * 8;
  const _Float16* pa1 = pa0 + 16 * DIM;
  const _Float16* pb0 = Bf + (size_t)(colBase + lm) * DIM + laneHi * 8;
  const _Float16* pb1 = pb0 + 16 * DIM;

  int gj0 = colBase + lm, gj1 = gj0 + 16;
  int rj0 = 0, rj1 = 0;
  int ri0[8], ri1[8];
  if (IS_G) {
    rj0 = rnk[gj0];
    rj1 = rnk[gj1];
#pragma unroll
    for (int r = 0; r < 8; ++r) {
      ri0[r] = rnk[rowBase + laneHi * 8 + r];
      ri1[r] = rnk[rowBase + 16 + laneHi * 8 + r];
    }
    // Warm the scattered u_G lines (rank-space gather) while the GEMM runs.
#pragma unroll
    for (int r = 0; r < 8; ++r) {
      __builtin_prefetch(U + (size_t)ri0[r] * N_PTS + rj0, 0, 3);
      __builtin_prefetch(U + (size_t)ri0[r] * N_PTS + rj1, 0, 3);
      __builtin_prefetch(U + (size_t)ri1[r] * N_PTS + rj0, 0, 3);
      __builtin_prefetch(U + (size_t)ri1[r] * N_PTS + rj1, 0, 3);
    }
  } else {
#pragma unroll
    for (int r = 0; r < 8; ++r) {
      __builtin_prefetch(U + (size_t)(rowBase + laneHi * 8 + r) * N_PTS + gj0, 0, 3);
      __builtin_prefetch(U + (size_t)(rowBase + 16 + laneHi * 8 + r) * N_PTS + gj0, 0, 3);
    }
  }

  v8f c00 = {}, c01 = {}, c10 = {}, c11 = {};
#pragma unroll
  for (int kk = 0; kk < DIM; kk += 32) {
    v16h a0 = load_frag_a(pa0 + kk);
    v16h a1 = load_frag_a(pa1 + kk);
    v16h b0 = load_frag_b(pb0 + kk);
    v16h b1 = load_frag_b(pb1 + kk);
    c00 = wmma16(a0, b0, c00);
    c01 = wmma16(a0, b1, c01);
    c10 = wmma16(a1, b0, c10);
    c11 = wmma16(a1, b1, c11);
  }

  float nb0v = nB[gj0];
  float nb1v = nB[gj1];
#pragma unroll
  for (int r = 0; r < 8; ++r) {
    // C/D layout: VGPR r -> M = r + 8*laneHi, N = lane&15
    int gi0 = rowBase + laneHi * 8 + r;
    int gi1 = gi0 + 16;
    float na0 = nA[gi0];
    float na1 = nA[gi1];
    float u00, u01, u10, u11;
    float m00 = 1.f, m01 = 1.f, m10 = 1.f, m11 = 1.f;
    if (IS_G) {
      u00 = U[(size_t)ri0[r] * N_PTS + rj0]; m00 = (ri0[r] < rj0) ? 1.f : 0.f;
      u01 = U[(size_t)ri0[r] * N_PTS + rj1]; m01 = (ri0[r] < rj1) ? 1.f : 0.f;
      u10 = U[(size_t)ri1[r] * N_PTS + rj0]; m10 = (ri1[r] < rj0) ? 1.f : 0.f;
      u11 = U[(size_t)ri1[r] * N_PTS + rj1]; m11 = (ri1[r] < rj1) ? 1.f : 0.f;
    } else {
      u00 = U[(size_t)gi0 * N_PTS + gj0];
      u01 = U[(size_t)gi0 * N_PTS + gj1];
      u10 = U[(size_t)gi1 * N_PTS + gj0];
      u11 = U[(size_t)gi1 * N_PTS + gj1];
    }
    out[(size_t)gi0 * N_PTS + gj0] = m00 * epi(c00[r], na0, nb0v, msc, u00);
    out[(size_t)gi0 * N_PTS + gj1] = m01 * epi(c01[r], na0, nb1v, msc, u01);
    out[(size_t)gi1 * N_PTS + gj0] = m10 * epi(c10[r], na1, nb0v, msc, u10);
    out[(size_t)gi1 * N_PTS + gj1] = m11 * epi(c11[r], na1, nb1v, msc, u11);
  }
}

extern "C" void kernel_launch(void* const* d_in, const int* in_sizes, int n_in,
                              void* d_out, int out_size, void* d_ws, size_t ws_size,
                              hipStream_t stream) {
  const float* uR   = (const float*)d_in[0];   // [4096,256]
  const float* uM   = (const float*)d_in[1];   // [4096,256]
  const float* gls  = (const float*)d_in[2];   // [1]
  const float* u_G  = (const float*)d_in[3];   // [4096,4096]
  const float* u_A  = (const float*)d_in[4];   // [4096,4096]
  float* out = (float*)d_out;

  // Workspace layout
  char* ws = (char*)d_ws;
  _Float16* uR16 = (_Float16*)ws;                                   // 2 MB
  _Float16* uM16 = (_Float16*)(ws + (size_t)N_PTS * DIM * 2);       // 2 MB
  float* nR   = (float*)(ws + 2 * (size_t)N_PTS * DIM * 2);
  float* nM   = nR + N_PTS;
  float* lcdf = nM + N_PTS;
  int*   rnk  = (int*)(lcdf + N_PTS);

  // 1) per-row stats + f16 conversion
  rowstats_kernel<<<N_PTS, DIM, 0, stream>>>(uR, uR16, nR, lcdf);
  rowstats_kernel<<<N_PTS, DIM, 0, stream>>>(uM, uM16, nM, nullptr);

  // 2) rank = inverse argsort of log-CDF (stable ties)
  rank_kernel<<<N_PTS / 256, 256, 0, stream>>>(lcdf, rnk);

  // 3) fused WMMA distance + sampling
  dim3 grid(N_PTS / 128, N_PTS / 64);
  // G: uR vs uR, u_G gathered in rank space, strict-upper mask, out[0]
  pair_kernel<true><<<grid, 256, 0, stream>>>(uR16, uR16, nR, nR, rnk, u_G, gls, out);
  // A: uM vs uR, u_A direct, out[1]
  pair_kernel<false><<<grid, 256, 0, stream>>>(uM16, uR16, nM, nR, nullptr, u_A, gls,
                                               out + (size_t)N_PTS * N_PTS);
}
